// InfiniAttentionLayer_45457933861048
// MI455X (gfx1250) — compile-verified, matrix operand order checked by reading, not run on hardware
//
#include <hip/hip_runtime.h>
#include <hip/hip_bf16.h>

// ---------------------------------------------------------------------------
// InfiniAttention for MI455X (gfx1250): bf16 WMMA (16x16x32) everywhere,
// projection GEMMs staged via CDNA5 async-to-LDS DMA (ASYNCcnt), and the
// memory-update GEMM staged via the Tensor Data Mover (TENSORcnt).
// ---------------------------------------------------------------------------

typedef __bf16 bf16_t;
typedef bf16_t bf16x16 __attribute__((ext_vector_type(16)));
typedef float  f32x8   __attribute__((ext_vector_type(8)));
typedef unsigned int u32x4 __attribute__((ext_vector_type(4)));
typedef int          i32x4 __attribute__((ext_vector_type(4)));
typedef int          i32x8 __attribute__((ext_vector_type(8)));

#define DIM_ 1024
#define H_   16
#define HD_  64
#define B_   2
#define S_   2048
#define MS_  (B_ * S_)   // 4096 rows in the flattened projection GEMMs

#if defined(__has_builtin)
#if __has_builtin(__builtin_amdgcn_tensor_load_to_lds)
#define USE_TDM 1
#endif
#endif

__device__ __forceinline__ f32x8 wmma_bf16(bf16x16 a, bf16x16 b, f32x8 c) {
  // D = A(16x32) * B(32x16) + C(16x16), fp32 accumulate
  return __builtin_amdgcn_wmma_f32_16x16x32_bf16(false, a, false, b,
                                                 (short)0, c, false, false);
}

__device__ __forceinline__ bf16x16 frag_ld(const bf16_t* p) {
  return *(const bf16x16*)p;   // 32B per lane; all strides keep 32B alignment
}

__device__ __forceinline__ f32x8 fzero8() {
  f32x8 z = {0.f, 0.f, 0.f, 0.f, 0.f, 0.f, 0.f, 0.f};
  return z;
}

// ---- CDNA5 async DMA: global -> LDS, 16 bytes, tracked by ASYNCcnt --------
__device__ __forceinline__ void async_ld_b128(unsigned lds_off, const void* gptr) {
  asm volatile("global_load_async_to_lds_b128 %0, %1, off"
               :: "v"(lds_off), "v"(gptr) : "memory");
}
__device__ __forceinline__ unsigned lds_off_of(const void* p) {
  // flat LDS aperture: addr[31:0] is the wave-relative LDS byte address
  return (unsigned)(unsigned long long)p;
}

// Stage one 64x32-bf16 tile (row stride in elements) into LDS.
// 128 threads: tid/2 = row, tid&1 = 32-byte half of the 64-byte row.
__device__ __forceinline__ void stage_tile_64x32(const bf16_t* g, int row_stride,
                                                 unsigned lds_base, int tid) {
  int row = tid >> 1;
  int seg = tid & 1;
  const bf16_t* gp = g + row * row_stride + seg * 16;
  unsigned l = lds_base + (unsigned)(row * 64 + seg * 32);  // byte offsets
  async_ld_b128(l, gp);
  async_ld_b128(l + 16, gp + 8);
}

#ifdef USE_TDM
// ---- Tensor Data Mover: DMA a 2D tile (tile_d0 x tile_d1 bf16 elements,
// dim0 contiguous, dim1 stride = stride0 elements) from global to LDS.
// Descriptor packing per CDNA5 ISA 8.3-8.5 (D# groups 0/1; groups 2/3 zero
// for tensors up to 2D). data_size=1 -> 2-byte elements.
// This toolchain exposes the 6-argument builtin form (extra i32x8, zeroed).
__device__ __forceinline__ void tdm_load_2d(unsigned lds_addr, const void* gaddr,
                                            unsigned tile_d0, unsigned tile_d1,
                                            unsigned tensor_d0, unsigned tensor_d1,
                                            unsigned long long stride0_elems) {
  unsigned long long ga = (unsigned long long)gaddr;
  u32x4 g0;
  g0[0] = 1u;                                   // count=1, user descriptor
  g0[1] = lds_addr;                             // LDS byte address
  g0[2] = (unsigned)(ga & 0xffffffffu);         // global_addr[31:0]
  g0[3] = (unsigned)((ga >> 32) & 0x01ffffffu)  // global_addr[56:32]
          | 0x80000000u;                        // type=2 ("image")
  i32x8 g1;
  g1[0] = (int)(1u << 16);                                  // data_size=2B
  g1[1] = (int)((tensor_d0 & 0xffffu) << 16);               // tensor_dim0[15:0]
  g1[2] = (int)((tensor_d0 >> 16) | ((tensor_d1 & 0xffffu) << 16));
  g1[3] = (int)((tensor_d1 >> 16) | ((tile_d0 & 0xffffu) << 16)); // tile_dim0
  g1[4] = (int)(tile_d1 & 0xffffu);                         // tile_dim1 (dim2=0)
  g1[5] = (int)(stride0_elems & 0xffffffffu);               // dim0_stride[31:0]
  g1[6] = (int)((stride0_elems >> 32) & 0xffffu);           // dim0_stride[47:32]
  g1[7] = 0;                                                // dim1_stride unused
  i32x4 z4 = {0, 0, 0, 0};
  i32x8 z8 = {0, 0, 0, 0, 0, 0, 0, 0};
  __builtin_amdgcn_tensor_load_to_lds(g0, g1, z4, z4, z8, 0);
}
#endif

// --------------------------- cast / transpose kernels ----------------------

__global__ void cast_f32_to_bf16(const float* __restrict__ src,
                                 bf16_t* __restrict__ dst, int n) {
  int i = blockIdx.x * blockDim.x + threadIdx.x;
  if (i < n) dst[i] = (bf16_t)src[i];
}

__global__ void mem_transpose_bf16(const float* __restrict__ memory,
                                   bf16_t* __restrict__ memT) {
  // memT[b,h,e,d] = memory[b,h,d,e]  (B operand for q_mem @ memory)
  int i = blockIdx.x * blockDim.x + threadIdx.x;
  if (i < B_ * H_ * HD_ * HD_) {
    int e  = i & (HD_ - 1);
    int d  = (i >> 6) & (HD_ - 1);
    int bh = i >> 12;
    memT[(bh * HD_ + e) * HD_ + d] = (bf16_t)memory[(bh * HD_ + d) * HD_ + e];
  }
}

// --------------------------- fused QKV projection --------------------------
// y = x @ W.T.  blockIdx.z: 0=Q, 1=K, 2=V.  Block = 64x64 tile, 4 waves.
// A/B K-slices (64x32) double-buffered in LDS via async DMA.

__global__ void __launch_bounds__(128)
qkv_gemm_kernel(const bf16_t* __restrict__ xb, const bf16_t* __restrict__ wb,
                bf16_t* __restrict__ qb, bf16_t* __restrict__ kb,
                bf16_t* __restrict__ vT, bf16_t* __restrict__ kmT) {
  __shared__ __align__(32) bf16_t At[2][64][32];
  __shared__ __align__(32) bf16_t Bt[2][64][32];

  const int tid  = threadIdx.x;
  const int lane = tid & 31;
  const int wave = tid >> 5;
  const int lr   = lane & 15;
  const int half = lane >> 4;
  const int m0 = blockIdx.x * 64 + wave * 16;
  const int n0 = blockIdx.y * 64;
  const int which = blockIdx.z;
  const bf16_t* W = wb + which * (DIM_ * DIM_);

  const bf16_t* Ab = xb + (size_t)(blockIdx.x * 64) * DIM_;
  const bf16_t* Bb = W + (size_t)n0 * DIM_;
  const unsigned ldsA[2] = { lds_off_of(&At[0][0][0]), lds_off_of(&At[1][0][0]) };
  const unsigned ldsB[2] = { lds_off_of(&Bt[0][0][0]), lds_off_of(&Bt[1][0][0]) };

  f32x8 acc[4];
#pragma unroll
  for (int j = 0; j < 4; ++j) acc[j] = fzero8();

  stage_tile_64x32(Ab, DIM_, ldsA[0], tid);
  stage_tile_64x32(Bb, DIM_, ldsB[0], tid);

  const int NIT = DIM_ / 32;
  for (int it = 0; it < NIT; ++it) {
    const int buf = it & 1;
    if (it + 1 < NIT) {
      stage_tile_64x32(Ab + (it + 1) * 32, DIM_, ldsA[buf ^ 1], tid);
      stage_tile_64x32(Bb + (it + 1) * 32, DIM_, ldsB[buf ^ 1], tid);
      asm volatile("s_wait_asynccnt 4" ::: "memory");  // current slice landed
    } else {
      asm volatile("s_wait_asynccnt 0" ::: "memory");
    }
    __syncthreads();

    bf16x16 af = frag_ld(&At[buf][wave * 16 + lr][half * 16]);
#pragma unroll
    for (int j = 0; j < 4; ++j)
      acc[j] = wmma_bf16(af, frag_ld(&Bt[buf][j * 16 + lr][half * 16]), acc[j]);

    __syncthreads();  // all waves done reading buf before it is re-filled
  }

#pragma unroll
  for (int j = 0; j < 4; ++j) {
#pragma unroll
    for (int r = 0; r < 8; ++r) {
      float v  = acc[j][r];
      int row  = m0 + r + 8 * half;          // flattened b*S + s
      int col  = n0 + j * 16 + lr;           // h*HD + d
      if (which == 0) {
        qb[row * DIM_ + col] = (bf16_t)v;
      } else if (which == 1) {
        kb[row * DIM_ + col] = (bf16_t)v;
        int b = row >> 11, s = row & (S_ - 1);
        int h = col >> 6,  d = col & (HD_ - 1);
        float km = v > 0.f ? v + 1.f : __expf(v);   // elu(k)+1
        kmT[((b * H_ + h) * HD_ + d) * S_ + s] = (bf16_t)km;
      } else {
        int b = row >> 11, s = row & (S_ - 1);
        int h = col >> 6,  e = col & (HD_ - 1);
        vT[((b * H_ + h) * HD_ + e) * S_ + s] = (bf16_t)v;  // v transposed
      }
    }
  }
}

// ------------------- causal attention + memory read + gate -----------------
// Block: one (b, h, 64-query tile). 4 waves, wave handles 16 query rows.

__global__ void __launch_bounds__(128)
attn_kernel(const bf16_t* __restrict__ qb, const bf16_t* __restrict__ kb,
            const bf16_t* __restrict__ vT, const bf16_t* __restrict__ memT,
            const float* __restrict__ norm, const float* __restrict__ gate,
            bf16_t* __restrict__ cb) {
  __shared__ __align__(32) bf16_t Plds[4][16][64];  // per-wave P transpose
  __shared__ float nrm_row[64];

  const int lane = threadIdx.x & 31;
  const int wave = threadIdx.x >> 5;
  const int lr   = lane & 15;
  const int half = lane >> 4;
  const int qblk = blockIdx.x;     // 0..31
  const int h    = blockIdx.y;
  const int b    = blockIdx.z;
  const int bh   = b * H_ + h;
  const float g     = 1.f / (1.f + __expf(-gate[h]));
  const float scale = 0.125f;      // HD^-0.5

  // per-row memory-read normalizer: sum_d (elu(q)+1) * norm[d]  (+1e-6)
  if (threadIdx.x < 64) {
    int s = qblk * 64 + threadIdx.x;
    const bf16_t* qrow = qb + (b * S_ + s) * DIM_ + h * HD_;
    const float*  nr   = norm + bh * HD_;
    float acc = 0.f;
    for (int d = 0; d < HD_; ++d) {
      float qv = (float)qrow[d];
      float qm = qv > 0.f ? qv + 1.f : __expf(qv);
      acc += qm * nr[d];
    }
    nrm_row[threadIdx.x] = acc + 1e-6f;
  }
  __syncthreads();

  const int qt = qblk * 64 + wave * 16;   // first query row of this wave

  // q A-fragments over HD=64 (two K=32 chunks)
  const bf16_t* qbase = qb + (b * S_ + qt + lr) * DIM_ + h * HD_ + half * 16;
  bf16x16 qa0 = frag_ld(qbase);
  bf16x16 qa1 = frag_ld(qbase + 32);

  // q_mem fragments: elu+1 applied elementwise in A-fragment layout
  bf16x16 qm0, qm1;
#pragma unroll
  for (int i = 0; i < 16; ++i) {
    float v0 = (float)qa0[i];
    qm0[i] = (bf16_t)(v0 > 0.f ? v0 + 1.f : __expf(v0));
    float v1 = (float)qa1[i];
    qm1[i] = (bf16_t)(v1 > 0.f ? v1 + 1.f : __expf(v1));
  }

  // compressive memory read: macc = q_mem @ memory  (B operand = memT rows)
  f32x8 macc[4];
  const bf16_t* mtb = memT + bh * HD_ * HD_;
#pragma unroll
  for (int j = 0; j < 4; ++j) {
    macc[j] = fzero8();
    const bf16_t* bp = mtb + (j * 16 + lr) * HD_ + half * 16;
    macc[j] = wmma_bf16(qm0, frag_ld(bp), macc[j]);
    macc[j] = wmma_bf16(qm1, frag_ld(bp + 32), macc[j]);
  }

  // flash-attention online softmax state (8 rows per lane-half)
  float m_i[8], l_i[8];
#pragma unroll
  for (int r = 0; r < 8; ++r) { m_i[r] = -3.0e38f; l_i[r] = 0.f; }
  f32x8 oacc[4];
#pragma unroll
  for (int j = 0; j < 4; ++j) oacc[j] = fzero8();

  for (int t = 0; t <= qblk; ++t) {
    // scores: S = q @ k^T  (k rows are contiguous -> direct B fragments)
    f32x8 sc[4];
    const bf16_t* kbase = kb + (b * S_ + t * 64 + lr) * DIM_ + h * HD_ + half * 16;
#pragma unroll
    for (int j = 0; j < 4; ++j) {
      sc[j] = fzero8();
      const bf16_t* bp = kbase + j * 16 * DIM_;
      sc[j] = wmma_bf16(qa0, frag_ld(bp), sc[j]);
      sc[j] = wmma_bf16(qa1, frag_ld(bp + 32), sc[j]);
    }

    // scale + causal mask + online softmax (row = r + 8*half across 16 lanes)
#pragma unroll
    for (int r = 0; r < 8; ++r) {
      int gm = qt + r + 8 * half;
      float sv[4];
      float rowmax = -3.0e38f;
#pragma unroll
      for (int j = 0; j < 4; ++j) {
        float x = sc[j][r] * scale;
        int gn = t * 64 + j * 16 + lr;
        if (gn > gm) x = -3.0e38f;
        sv[j] = x;
        rowmax = fmaxf(rowmax, x);
      }
      for (int m = 8; m >= 1; m >>= 1)
        rowmax = fmaxf(rowmax, __shfl_xor(rowmax, m, 32));
      float mnew = fmaxf(m_i[r], rowmax);
      float corr = __expf(m_i[r] - mnew);
      float rsum = 0.f;
#pragma unroll
      for (int j = 0; j < 4; ++j) {
        float p = __expf(sv[j] - mnew);
        rsum += p;
        Plds[wave][r + 8 * half][j * 16 + lr] = (bf16_t)p;  // transpose via LDS
      }
      for (int m = 8; m >= 1; m >>= 1) rsum += __shfl_xor(rsum, m, 32);
      l_i[r] = l_i[r] * corr + rsum;
      m_i[r] = mnew;
#pragma unroll
      for (int j = 0; j < 4; ++j) oacc[j][r] *= corr;
    }

    // O += P @ V   (P A-fragments from LDS; V columns from vT rows)
    const bf16_t* vbb = vT + bh * HD_ * S_ + t * 64 + half * 16;
#pragma unroll
    for (int kk2 = 0; kk2 < 2; ++kk2) {
      bf16x16 pf = frag_ld(&Plds[wave][lr][kk2 * 32 + half * 16]);
#pragma unroll
      for (int j = 0; j < 4; ++j) {
        const bf16_t* bp = vbb + (j * 16 + lr) * S_ + kk2 * 32;
        oacc[j] = wmma_bf16(pf, frag_ld(bp), oacc[j]);
      }
    }
  }

  // gated combine, write merged-head layout [B,S,DIM] bf16
#pragma unroll
  for (int j = 0; j < 4; ++j) {
#pragma unroll
    for (int r = 0; r < 8; ++r) {
      int blocal = wave * 16 + r + 8 * half;
      int s = qblk * 64 + blocal;
      float localv = oacc[j][r] / l_i[r];
      float memv   = macc[j][r] / nrm_row[blocal];
      float comb   = g * memv + (1.f - g) * localv;
      cb[(b * S_ + s) * DIM_ + h * HD_ + j * 16 + lr] = (bf16_t)comb;
    }
  }
}

// ------------------------- memory / norm update ----------------------------
// new_memory = memory + kmem^T @ v   (64x64, K = S = 2048, WMMA).
// K-slices (64x32) are DMA'd into LDS by the Tensor Data Mover (one wave
// issues; TENSORcnt tracks completion), double buffered.

__global__ void __launch_bounds__(128)
memupd_kernel(const bf16_t* __restrict__ kmT, const bf16_t* __restrict__ vT,
              const float* __restrict__ memory, const float* __restrict__ norm,
              float* __restrict__ new_memory, float* __restrict__ new_norm) {
  __shared__ __align__(32) bf16_t At[2][64][32];   // kmem^T slice
  __shared__ __align__(32) bf16_t Bt[2][64][32];   // v^T slice

  const int tid  = threadIdx.x;
  const int lane = tid & 31;
  const int wave = tid >> 5;
  const int lr   = lane & 15;
  const int half = lane >> 4;
  const int h = blockIdx.x, b = blockIdx.y;
  const int bh = b * H_ + h;

  const bf16_t* Aslab = kmT + (size_t)bh * HD_ * S_;   // [64][S_]
  const bf16_t* Bslab = vT + (size_t)bh * HD_ * S_;    // [64][S_]
  const unsigned ldsA[2] = { lds_off_of(&At[0][0][0]), lds_off_of(&At[1][0][0]) };
  const unsigned ldsB[2] = { lds_off_of(&Bt[0][0][0]), lds_off_of(&Bt[1][0][0]) };

  f32x8 acc[4];
#pragma unroll
  for (int j = 0; j < 4; ++j) acc[j] = fzero8();

  // prologue: slice 0 -> buffer 0
#ifdef USE_TDM
  if (wave == 0) {
    tdm_load_2d(ldsA[0], Aslab, 32, 64, S_, 64, S_);
    tdm_load_2d(ldsB[0], Bslab, 32, 64, S_, 64, S_);
  }
#else
  stage_tile_64x32(Aslab, S_, ldsA[0], tid);
  stage_tile_64x32(Bslab, S_, ldsB[0], tid);
#endif

  const int NIT = S_ / 32;
  for (int it = 0; it < NIT; ++it) {
    const int buf = it & 1;
    const bool more = (it + 1 < NIT);
#ifdef USE_TDM
    if (wave == 0) {
      if (more) {
        tdm_load_2d(ldsA[buf ^ 1], Aslab + (it + 1) * 32, 32, 64, S_, 64, S_);
        tdm_load_2d(ldsB[buf ^ 1], Bslab + (it + 1) * 32, 32, 64, S_, 64, S_);
        __builtin_amdgcn_s_wait_tensorcnt((short)2);   // current slice landed
      } else {
        __builtin_amdgcn_s_wait_tensorcnt((short)0);
      }
    }
#else
    if (more) {
      stage_tile_64x32(Aslab + (it + 1) * 32, S_, ldsA[buf ^ 1], tid);
      stage_tile_64x32(Bslab + (it + 1) * 32, S_, ldsB[buf ^ 1], tid);
      asm volatile("s_wait_asynccnt 4" ::: "memory");
    } else {
      asm volatile("s_wait_asynccnt 0" ::: "memory");
    }
#endif
    __syncthreads();

    bf16x16 af = frag_ld(&At[buf][wave * 16 + lr][half * 16]);
#pragma unroll
    for (int j = 0; j < 4; ++j)
      acc[j] = wmma_bf16(af, frag_ld(&Bt[buf][j * 16 + lr][half * 16]), acc[j]);

    __syncthreads();
  }

#pragma unroll
  for (int j = 0; j < 4; ++j) {
#pragma unroll
    for (int r = 0; r < 8; ++r) {
      int d = wave * 16 + r + 8 * half;
      int e = j * 16 + lr;
      int idx = (bh * HD_ + d) * HD_ + e;
      new_memory[idx] = memory[idx] + acc[j][r];
    }
  }

  if (threadIdx.x < 64) {
    int d = threadIdx.x;
    const bf16_t* kr = kmT + (bh * HD_ + d) * S_;
    float s = 0.f;
    for (int i = 0; i < S_; ++i) s += (float)kr[i];
    new_norm[bh * HD_ + d] = norm[bh * HD_ + d] + s;
  }
}

// ------------------------- output projection -------------------------------
// out = combined @ Wo.T, fp32 result; async-LDS double-buffer scheme.

__global__ void __launch_bounds__(128)
oproj_kernel(const bf16_t* __restrict__ cb, const bf16_t* __restrict__ Wob,
             float* __restrict__ out) {
  __shared__ __align__(32) bf16_t At[2][64][32];
  __shared__ __align__(32) bf16_t Bt[2][64][32];

  const int tid  = threadIdx.x;
  const int lane = tid & 31;
  const int wave = tid >> 5;
  const int lr   = lane & 15;
  const int half = lane >> 4;
  const int m0 = blockIdx.x * 64 + wave * 16;
  const int n0 = blockIdx.y * 64;

  const bf16_t* Ab = cb + (size_t)(blockIdx.x * 64) * DIM_;
  const bf16_t* Bb = Wob + (size_t)n0 * DIM_;
  const unsigned ldsA[2] = { lds_off_of(&At[0][0][0]), lds_off_of(&At[1][0][0]) };
  const unsigned ldsB[2] = { lds_off_of(&Bt[0][0][0]), lds_off_of(&Bt[1][0][0]) };

  f32x8 acc[4];
#pragma unroll
  for (int j = 0; j < 4; ++j) acc[j] = fzero8();

  stage_tile_64x32(Ab, DIM_, ldsA[0], tid);
  stage_tile_64x32(Bb, DIM_, ldsB[0], tid);

  const int NIT = DIM_ / 32;
  for (int it = 0; it < NIT; ++it) {
    const int buf = it & 1;
    if (it + 1 < NIT) {
      stage_tile_64x32(Ab + (it + 1) * 32, DIM_, ldsA[buf ^ 1], tid);
      stage_tile_64x32(Bb + (it + 1) * 32, DIM_, ldsB[buf ^ 1], tid);
      asm volatile("s_wait_asynccnt 4" ::: "memory");
    } else {
      asm volatile("s_wait_asynccnt 0" ::: "memory");
    }
    __syncthreads();

    bf16x16 af = frag_ld(&At[buf][wave * 16 + lr][half * 16]);
#pragma unroll
    for (int j = 0; j < 4; ++j)
      acc[j] = wmma_bf16(af, frag_ld(&Bt[buf][j * 16 + lr][half * 16]), acc[j]);

    __syncthreads();
  }

#pragma unroll
  for (int j = 0; j < 4; ++j)
#pragma unroll
    for (int r = 0; r < 8; ++r)
      out[(m0 + r + 8 * half) * DIM_ + n0 + j * 16 + lr] = acc[j][r];
}

// ------------------------------- launch ------------------------------------

extern "C" void kernel_launch(void* const* d_in, const int* in_sizes, int n_in,
                              void* d_out, int out_size, void* d_ws, size_t ws_size,
                              hipStream_t stream) {
  const float* x      = (const float*)d_in[0];   // [B,S,DIM]
  const float* memory = (const float*)d_in[1];   // [B,H,HD,HD]
  const float* norm   = (const float*)d_in[2];   // [B,H,HD]
  const float* Wq     = (const float*)d_in[3];
  const float* Wk     = (const float*)d_in[4];
  const float* Wv     = (const float*)d_in[5];
  const float* Wo     = (const float*)d_in[6];
  const float* gate   = (const float*)d_in[7];   // [1,H,1,1]

  float* out        = (float*)d_out;                      // B*S*DIM
  float* new_memory = out + (size_t)B_ * S_ * DIM_;       // B*H*HD*HD
  float* new_norm   = new_memory + (size_t)B_ * H_ * HD_ * HD_;

  // workspace carve-up (bf16 buffers), ~56.3 MB total
  char* ws = (char*)d_ws;
  const size_t SZ_XB = (size_t)MS_ * DIM_ * sizeof(bf16_t);      // 8 MB
  const size_t SZ_W  = (size_t)DIM_ * DIM_ * sizeof(bf16_t);     // 2 MB
  bf16_t* xb  = (bf16_t*)ws;                       ws += SZ_XB;
  bf16_t* wb  = (bf16_t*)ws;                       ws += 4 * SZ_W;  // q,k,v,o
  bf16_t* qb  = (bf16_t*)ws;                       ws += SZ_XB;
  bf16_t* kb  = (bf16_t*)ws;                       ws += SZ_XB;
  bf16_t* vT  = (bf16_t*)ws;                       ws += SZ_XB;
  bf16_t* kmT = (bf16_t*)ws;                       ws += SZ_XB;
  bf16_t* cb  = (bf16_t*)ws;                       ws += SZ_XB;
  bf16_t* memT = (bf16_t*)ws;                      ws += (size_t)B_ * H_ * HD_ * HD_ * sizeof(bf16_t);

  // 1) casts
  {
    int n = MS_ * DIM_;
    cast_f32_to_bf16<<<(n + 255) / 256, 256, 0, stream>>>(x, xb, n);
    int nw = DIM_ * DIM_;
    cast_f32_to_bf16<<<(nw + 255) / 256, 256, 0, stream>>>(Wq, wb + 0 * nw, nw);
    cast_f32_to_bf16<<<(nw + 255) / 256, 256, 0, stream>>>(Wk, wb + 1 * nw, nw);
    cast_f32_to_bf16<<<(nw + 255) / 256, 256, 0, stream>>>(Wv, wb + 2 * nw, nw);
    cast_f32_to_bf16<<<(nw + 255) / 256, 256, 0, stream>>>(Wo, wb + 3 * nw, nw);
    int nm = B_ * H_ * HD_ * HD_;
    mem_transpose_bf16<<<(nm + 255) / 256, 256, 0, stream>>>(memory, memT);
  }

  // 2) fused QKV projection (z: 0=Q,1=K,2=V)
  qkv_gemm_kernel<<<dim3(MS_ / 64, DIM_ / 64, 3), 128, 0, stream>>>(
      xb, wb, qb, kb, vT, kmT);

  // 3) causal attention + memory read + gated combine
  attn_kernel<<<dim3(S_ / 64, H_, B_), 128, 0, stream>>>(
      qb, kb, vT, memT, norm, gate, cb);

  // 4) memory + norm update (TDM-staged)
  memupd_kernel<<<dim3(H_, B_), 128, 0, stream>>>(
      kmT, vT, memory, norm, new_memory, new_norm);

  // 5) output projection
  oproj_kernel<<<dim3(MS_ / 64, DIM_ / 64), 128, 0, stream>>>(cb, wb + 3 * DIM_ * DIM_, out);
}
